// CG_model_jit_48911087567271
// MI455X (gfx1250) — matrix operand order repaired and checked by reading, not compiled, not run on hardware
//
#include <hip/hip_runtime.h>
#include <hip/hip_bf16.h>

typedef __attribute__((ext_vector_type(16))) _Float16 v16h;
typedef __attribute__((ext_vector_type(8)))  float    v8f;

static constexpr int Nn  = 20000;
static constexpr int Ee  = 640000;

// fast math helpers: single-instruction v_rcp_f32 / v_sqrt_f32
static __device__ __forceinline__ float frcp(float x)  { return __builtin_amdgcn_rcpf(x); }
static __device__ __forceinline__ float fsqrt(float x) { return __builtin_amdgcn_sqrtf(x); }
static __device__ __forceinline__ float silu(float z)  { return z * frcp(1.f + __expf(-z)); }
static __device__ __forceinline__ float softplus(float z) {
    return (z > 20.f) ? z : __logf(1.f + __expf(z));
}

// ---------------------------------------------------------------------------
// zero accumulators (accV: 3N, accS: N) in workspace
// ---------------------------------------------------------------------------
__global__ void k_zero(float* acc, int n) {
    int t = blockIdx.x * blockDim.x + threadIdx.x;
    if (t < n) acc[t] = 0.0f;
}

// ---------------------------------------------------------------------------
// Node kernel: mono-MLP U at 4 perturbed inputs -> T, P, C per node, E_out
// ---------------------------------------------------------------------------
struct NodeParams {
    const float *v, *S, *d;
    const float *Uw1, *Ub1, *Uw2, *Ub2, *Uw3, *Ub3;
    const float *log_m;
    float *T, *P, *Cn, *Eout;
};

__global__ __launch_bounds__(256) void k_node(NodeParams p) {
    int n = blockIdx.x * blockDim.x + threadIdx.x;
    if (n >= Nn) return;
    const float EPSU = 0.01f;
    float S0 = p.S[n];
    float V0 = frcp(p.d[n]);
    float Ss[4] = {S0, S0 + EPSU, S0, S0 - EPSU};
    float Vs[4] = {V0, V0, V0 + EPSU, V0};
    float Uv[4];
#pragma unroll 1
    for (int q = 0; q < 4; ++q) {
        float Sx = Ss[q], Vx = Vs[q];
        float h2[64];
#pragma unroll
        for (int c = 0; c < 64; ++c) h2[c] = p.Ub2[c];
#pragma unroll 1
        for (int k = 0; k < 64; ++k) {
            float z = Sx * fabsf(p.Uw1[k]) - Vx * fabsf(p.Uw1[64 + k]) + p.Ub1[k];
            float a = softplus(z);
#pragma unroll
            for (int c = 0; c < 64; ++c) h2[c] += a * fabsf(p.Uw2[k * 64 + c]);
        }
        float u = p.Ub3[0];
#pragma unroll
        for (int c = 0; c < 64; ++c) u += softplus(h2[c]) * fabsf(p.Uw3[c]);
        Uv[q] = u;
    }
    float U = Uv[0], USp = Uv[1], UVp = Uv[2], USm = Uv[3];
    float T  = (USp - U) * 100.f;              // /EPSU
    float P  = -(UVp - U) * 100.f;
    float Cn = T * (EPSU * EPSU) * frcp(USp - 2.f * U + USm);
    p.T[n] = T; p.P[n] = P; p.Cn[n] = Cn;
    float m = __expf(p.log_m[0]);
    float vx = p.v[3 * n], vy = p.v[3 * n + 1], vz = p.v[3 * n + 2];
    p.Eout[n] = U + 0.5f * m * (vx * vx + vy * vy + vz * vz);
}

// ---------------------------------------------------------------------------
// Edge kernel: fused A/B/C/W MLPs via v_wmma_f32_16x16x32_f16 (16 edges/wave),
// per-edge physics epilogue, atomic scatter into node accumulators.
// ---------------------------------------------------------------------------
struct EdgeParams {
    const int*   eidx;
    const float* rij;
    const float* v;
    const float* d;
    const float* dW;
    const float* dV;
    const float* T;
    const float* P;
    const float* Cn;
    float*       accV;
    float*       accS;
    const float* log_kB;
    const float* log_m;
    const float* w[4][6];   // [A,B,C,W][w1,b1,w2,b2,w3,b3]
};

__global__ __launch_bounds__(256) void k_edge(EdgeParams p) {
    __shared__ _Float16 w2h[4 * 64 * 64];   // W2 as f16, [mlp][n][k] (B-frag friendly)
    __shared__ float    w1s[4 * 2 * 64];
    __shared__ float    b1s[4 * 64];
    __shared__ float    b2s[4 * 64];
    __shared__ float    w3s[4 * 64];
    __shared__ float    b3s[4];
    __shared__ float    scal[8 * 14 * 16];  // [wave][variant-slot][row]

    const int tid = threadIdx.x;

    // stage weights into LDS (transpose W2 to [n][k], convert to f16)
    for (int idx = tid; idx < 4 * 64 * 64; idx += 256) {
        int mlp = idx >> 12;
        int rem = idx & 4095;
        int n = rem >> 6, k = rem & 63;
        w2h[(mlp << 12) + (n << 6) + k] = (_Float16)p.w[mlp][2][k * 64 + n];
    }
    for (int idx = tid; idx < 4 * 64; idx += 256) {
        int mlp = idx >> 6, k = idx & 63;
        b1s[idx] = p.w[mlp][1][k];
        b2s[idx] = p.w[mlp][3][k];
        w3s[idx] = p.w[mlp][4][k];
        if (mlp < 3) {
            w1s[mlp * 128 + k]      = p.w[mlp][0][k];
            w1s[mlp * 128 + 64 + k] = p.w[mlp][0][64 + k];
        } else {
            w1s[3 * 128 + k]      = p.w[3][0][k];   // W mlp: 1 input row
            w1s[3 * 128 + 64 + k] = 0.0f;
        }
    }
    if (tid < 4) b3s[tid] = p.w[tid][5][0];
    __syncthreads();

    const int lane = tid & 31;
    const int wave = tid >> 5;
    const int row  = lane & 15;
    const int hiK  = (lane >= 16) ? 16 : 0;   // A-frag K offset (upper half-wave)
    const int hiB  = (lane >= 16) ? 16 : 0;   // B-frag K offset
    const int e    = blockIdx.x * 128 + wave * 16 + row;  // E == 128*gridDim exactly

    // per-row edge scalars (duplicated in both lane halves so A-frags are local)
    const int ni = p.eidx[e];
    const int nj = p.eidx[Ee + e];
    const float rx = p.rij[3 * e], ry = p.rij[3 * e + 1], rz = p.rij[3 * e + 2];
    const float r_ = fsqrt(rx * rx + ry * ry + rz * rz);
    const float Ti = p.T[ni], Tj = p.T[nj];

    float* myScal = &scal[wave * 14 * 16];

#pragma unroll 1
    for (int mlp = 0; mlp < 4; ++mlp) {
        const int nvar = (mlp == 3) ? 2 : 4;
        const float*    w1m = &w1s[mlp * 128];
        const float*    b1m = &b1s[mlp * 64];
        const float*    b2m = &b2s[mlp * 64];
        const float*    w3m = &w3s[mlp * 64];
        const _Float16* w2m = &w2h[mlp << 12];
#pragma unroll 1
        for (int var = 0; var < nvar; ++var) {
            float x0, x1;
            if (mlp == 3) {
                x0 = (var == 0) ? (r_ + 1e-3f) : (r_ - 1e-3f);  // Sk = r±eps (H=1)
                x1 = 0.0f;
            } else {
                x0 = r_;
                x1 = (var == 0) ? Ti : (var == 1) ? Tj
                   : (var == 2) ? (Ti + 1e-3f) : (Tj + 1e-3f);
            }

            // layer-1 (2->64, SiLU) directly into both 16x32 A-fragments
            v16h a0, a1;
#pragma unroll
            for (int u = 0; u < 8; ++u) {
#pragma unroll
                for (int pp = 0; pp < 2; ++pp) {
                    int kb = hiK + u * 2 + pp;
                    float z0 = x0 * w1m[kb] + x1 * w1m[64 + kb] + b1m[kb];
                    float z1 = x0 * w1m[32 + kb] + x1 * w1m[96 + kb] + b1m[32 + kb];
                    a0[2 * u + pp] = (_Float16)silu(z0);
                    a1[2 * u + pp] = (_Float16)silu(z1);
                }
            }

            // layer-2 + layer-3: per N-group, 2 chained WMMAs then immediate
            // SiLU(.+b2).w3 fold so only one v8f accumulator is ever live.
            float part[8];
#pragma unroll
            for (int rr = 0; rr < 8; ++rr) part[rr] = 0.f;
            const v8f zero = {0.f, 0.f, 0.f, 0.f, 0.f, 0.f, 0.f, 0.f};
#pragma unroll
            for (int g = 0; g < 4; ++g) {
                const _Float16* wp = &w2m[((g * 16 + row) << 6) + hiB];
                v16h b0 = *(const v16h*)wp;          // K 0..31 slice
                v16h b1 = *(const v16h*)(wp + 32);   // K 32..63 slice
                v8f acc = __builtin_amdgcn_wmma_f32_16x16x32_f16(
                    false, a0, false, b0, (short)0, zero, false, false);
                acc = __builtin_amdgcn_wmma_f32_16x16x32_f16(
                    false, a1, false, b1, (short)0, acc, false, false);
                int nIdx = g * 16 + row;
                float b2v = b2m[nIdx], w3v = w3m[nIdx];
#pragma unroll
                for (int rr = 0; rr < 8; ++rr) {
                    part[rr] += silu(acc[rr] + b2v) * w3v;
                }
            }
#pragma unroll
            for (int rr = 0; rr < 8; ++rr) {
                float s = part[rr];
                s += __shfl_xor(s, 1);
                s += __shfl_xor(s, 2);
                s += __shfl_xor(s, 4);
                s += __shfl_xor(s, 8);
                part[rr] = s + b3s[mlp];
            }
            int slot = (mlp < 3) ? (mlp * 4 + var) : (12 + var);
            if (row == 0) {  // lanes 0 (rows 0..7) and 16 (rows 8..15)
                int base = slot * 16 + ((lane >= 16) ? 8 : 0);
#pragma unroll
                for (int rr = 0; rr < 8; ++rr) myScal[base + rr] = part[rr];
            }
            // fence scheduling: stop cross-variant software pipelining that
            // pushes live ranges past the 256-VGPR cliff (MSB addressing tax)
            __builtin_amdgcn_sched_barrier(0);
        }
    }
    __syncthreads();

    // -------- per-edge physics epilogue (one edge per lane 0..15) --------
    if (lane < 16) {
        const float k_B   = __expf(p.log_kB[0]);
        const float invm  = __expf(-p.log_m[0]);     // 1/m
        const float sq2kB = fsqrt(2.f * k_B);
        const float inv_sqdt = 10.f;   // 1/sqrt(DT), DT=0.01

        const float inv_r = frcp(r_ + 1e-8f);
        const float ex = rx * inv_r, ey = ry * inv_r, ez = rz * inv_r;

        const float* sc = myScal;
        const float A_i  = sc[0 * 16 + row],  A_j  = sc[1 * 16 + row];
        const float A_ie = sc[2 * 16 + row],  A_je = sc[3 * 16 + row];
        const float B_i  = sc[4 * 16 + row],  B_j  = sc[5 * 16 + row];
        const float B_ie = sc[6 * 16 + row],  B_je = sc[7 * 16 + row];
        const float Ce_i = sc[8 * 16 + row],  Ce_j = sc[9 * 16 + row];
        const float Ce_ie= sc[10 * 16 + row], Ce_je= sc[11 * 16 + row];
        const float oWp  = sc[12 * 16 + row], oWm  = sc[13 * 16 + row];

        // kernel-gradient from W-MLP finite difference
        const float SkP = r_ + 1e-3f, SkM = r_ - 1e-3f;
        const float WpP = __expf(oWp) * (1.f - SkP * SkP);
        const float WpM = __expf(oWm) * (1.f - SkM * SkM);
        const float dW_dr = (WpP - WpM) * 500.f;   // /(2*1e-3)

        const float A_ij = A_i * A_j, Bp_ij = B_i * B_j, Cp_ij = Ce_i * Ce_j;
        const float invET = 1000.f;  // 1/eps_T
        const float gA_Ti = 2.f * A_ij * (A_ie * A_j - A_ij) * invET;
        const float gB_Ti = 2.f * Bp_ij * (B_ie * B_j - Bp_ij) * invET;
        const float gC_Ti = 2.f * Cp_ij * (Ce_ie * Ce_j - Cp_ij) * invET;
        const float gA_Tj = 2.f * A_ij * (A_i * A_je - A_ij) * invET;
        const float gB_Tj = 2.f * Bp_ij * (B_i * B_je - Bp_ij) * invET;
        const float gC_Tj = 2.f * Cp_ij * (Ce_i * Ce_je - Cp_ij) * invET;

        const float vix = p.v[3 * ni], viy = p.v[3 * ni + 1], viz = p.v[3 * ni + 2];
        const float vjx = p.v[3 * nj], vjy = p.v[3 * nj + 1], vjz = p.v[3 * nj + 2];
        const float dvx = vix - vjx, dvy = viy - vjy, dvz = viz - vjz;
        const float ev  = ex * dvx + ey * dvy + ez * dvz;
        const float vsq = dvx * dvx + dvy * dvy + dvz * dvz;

        const float di = p.d[ni], dj = p.d[nj];
        const float Pi = p.P[ni], Pj = p.P[nj];
        const float invTi = frcp(Ti), invTj = frcp(Tj);
        const float invCi = frcp(p.Cn[ni]), invCj = frcp(p.Cn[nj]);

        const float cPd = Pi * frcp(di * di) + Pj * frcp(dj * dj);

        const float A2 = A_ij * A_ij, B2 = Bp_ij * Bp_ij;
        const float A2h = 0.5f * A2;
        const float third = 1.f / 3.f;
        const float coefE = A2h + (B2 - A2) * third;

        const float sT = invTi + invTj;
        const float c1 = -(invTi * invCi + invTj * invCj);
        const float a2i = 0.5f * gA_Ti, e2i = a2i + (gB_Ti - gA_Ti) * third;
        const float a2j = 0.5f * gA_Tj, e2j = a2j + (gB_Tj - gA_Tj) * third;

        // combined antisymmetric velocity contribution (i:+, j:-)
        const float cv = (-0.5f * sT * A2h   - 0.5f * k_B * (c1 * A2h   + a2i * invCi + a2j * invCj)) * invm;
        const float ce = (-0.5f * sT * coefE - 0.5f * k_B * (c1 * coefE + e2i * invCi + e2j * invCj)) * invm;

        // stochastic tensor term:  M_e . e_ij
        const float* dWe = &p.dW[9 * e];
        const float d00 = dWe[0], d01 = dWe[1], d02 = dWe[2];
        const float d10 = dWe[3], d11 = dWe[4], d12 = dWe[5];
        const float d20 = dWe[6], d21 = dWe[7], d22 = dWe[8];
        const float tr  = d00 + d11 + d22;
        const float trD = tr * third;
        const float sxx = d00 - trD, syy = d11 - trD, szz = d22 - trD;
        const float sxy = 0.5f * (d01 + d10), sxz = 0.5f * (d02 + d20), syz = 0.5f * (d12 + d21);
        const float bxv = sxx * ex + sxy * ey + sxz * ez;
        const float byv = sxy * ex + syy * ey + syz * ez;
        const float bzv = sxz * ex + syz * ey + szz * ez;
        const float tmx = A_ij * bxv + Bp_ij * trD * ex;
        const float tmy = A_ij * byv + Bp_ij * trD * ey;
        const float tmz = A_ij * bzv + Bp_ij * trD * ez;

        const float ctil   = sq2kB * invm * inv_sqdt;
        const float cegrad = ce * ev - cPd * dW_dr * invm;
        const float cvx = cv * dvx + cegrad * ex + ctil * tmx;
        const float cvy = cv * dvy + cegrad * ey + ctil * tmy;
        const float cvz = cv * dvz + cegrad * ez + ctil * tmz;

        atomicAdd(&p.accV[3 * ni + 0],  cvx);
        atomicAdd(&p.accV[3 * ni + 1],  cvy);
        atomicAdd(&p.accV[3 * ni + 2],  cvz);
        atomicAdd(&p.accV[3 * nj + 0], -cvx);
        atomicAdd(&p.accV[3 * nj + 1], -cvy);
        atomicAdd(&p.accV[3 * nj + 2], -cvz);

        // entropy channel (node-side /T applied in k_final)
        const float aux2 = (A2h * vsq + coefE * ev * ev) * 0.25f;
        const float C2   = Cp_ij * Cp_ij;
        const float MgS_i = sT * aux2 + (invTi - invTj) * C2;
        const float MgS_j = sT * aux2 + (invTj - invTi) * C2;
        const float t1s = -(2.f * invTi * invCi + invTj * invCj) * aux2;
        const float t2s = (a2i * vsq + e2i * ev * ev) * invCi * 0.25f
                        + (a2j * vsq + e2j * ev * ev) * invCj * 0.25f;
        const float t4s = -(2.f * invTi * invCi - invTj * invCj) * C2;
        const float t5s = gC_Ti * invCi - gC_Tj * invCj;
        const float t6s = -(2.f * A2 + (B2 - A2) * third);  // (D+1)/2 = 2 for D=3
        const float dMS_i = t1s + t2s + t4s + t5s + t6s * invm;
        const float dMS_j = t1s + t2s - t4s - t5s + t6s * invm;
        const float termS = -0.5f * (tmx * dvx + tmy * dvy + tmz * dvz);
        const float dVe = p.dV[e];
        const float ct2 = sq2kB * inv_sqdt;
        const float s_i = (MgS_i + k_B * dMS_i) + ct2 * (termS + Cp_ij * dVe);
        const float s_j = (MgS_j + k_B * dMS_j) + ct2 * (termS - Cp_ij * dVe);
        atomicAdd(&p.accS[ni], s_i);
        atomicAdd(&p.accS[nj], s_j);
    }
}

// ---------------------------------------------------------------------------
// finalize: dvdt -> out[0:3N), dSdt = accS/T -> out[3N:4N)  (E_out written by k_node)
// ---------------------------------------------------------------------------
__global__ void k_final(const float* accV, const float* accS, const float* T,
                        float* outV, float* outS) {
    int n = blockIdx.x * blockDim.x + threadIdx.x;
    if (n >= Nn) return;
    outV[3 * n + 0] = accV[3 * n + 0];
    outV[3 * n + 1] = accV[3 * n + 1];
    outV[3 * n + 2] = accV[3 * n + 2];
    outS[n] = accS[n] * frcp(T[n]);
}

// ---------------------------------------------------------------------------
extern "C" void kernel_launch(void* const* d_in, const int* in_sizes, int n_in,
                              void* d_out, int out_size, void* d_ws, size_t ws_size,
                              hipStream_t stream) {
    (void)in_sizes; (void)n_in; (void)out_size; (void)ws_size;
    const float* v    = (const float*)d_in[0];
    const int*   eidx = (const int*)  d_in[1];
    const float* rij  = (const float*)d_in[2];
    const float* S    = (const float*)d_in[3];
    const float* dcty = (const float*)d_in[4];
    const float* dW   = (const float*)d_in[5];
    const float* dV   = (const float*)d_in[6];
    // params pytree, sorted-key flatten: A(7..12), B(13..18), C(19..24),
    // U(25..30), W(31..36), log_k_B(37), log_m(38); each MLP = w1,b1,w2,b2,w3,b3
    const int baseA = 7, baseB = 13, baseC = 19, baseU = 25, baseW = 31;
    const float* log_kB = (const float*)d_in[37];
    const float* log_m  = (const float*)d_in[38];

    float* ws   = (float*)d_ws;
    float* accV = ws;                 // 3N
    float* accS = ws + 3 * Nn;        // N
    float* Tn   = ws + 4 * Nn;        // N
    float* Pn   = ws + 5 * Nn;        // N
    float* Cnn  = ws + 6 * Nn;        // N

    float* out  = (float*)d_out;
    float* outV = out;                // N x 3
    float* outS = out + 3 * Nn;       // N
    float* outE = out + 4 * Nn;       // N

    k_zero<<<(4 * Nn + 255) / 256, 256, 0, stream>>>(accV, 4 * Nn);

    NodeParams np;
    np.v = v; np.S = S; np.d = dcty;
    np.Uw1 = (const float*)d_in[baseU + 0];
    np.Ub1 = (const float*)d_in[baseU + 1];
    np.Uw2 = (const float*)d_in[baseU + 2];
    np.Ub2 = (const float*)d_in[baseU + 3];
    np.Uw3 = (const float*)d_in[baseU + 4];
    np.Ub3 = (const float*)d_in[baseU + 5];
    np.log_m = log_m;
    np.T = Tn; np.P = Pn; np.Cn = Cnn; np.Eout = outE;
    k_node<<<(Nn + 255) / 256, 256, 0, stream>>>(np);

    EdgeParams ep;
    ep.eidx = eidx; ep.rij = rij; ep.v = v; ep.d = dcty;
    ep.dW = dW; ep.dV = dV;
    ep.T = Tn; ep.P = Pn; ep.Cn = Cnn;
    ep.accV = accV; ep.accS = accS;
    ep.log_kB = log_kB; ep.log_m = log_m;
    const int bases[4] = {baseA, baseB, baseC, baseW};
    for (int mlp = 0; mlp < 4; ++mlp)
        for (int q = 0; q < 6; ++q)
            ep.w[mlp][q] = (const float*)d_in[bases[mlp] + q];
    k_edge<<<Ee / 128, 256, 0, stream>>>(ep);

    k_final<<<(Nn + 255) / 256, 256, 0, stream>>>(accV, accS, Tn, outV, outS);
}
